// kernel_generated_6_21887153341314
// MI455X (gfx1250) — matrix operand order repaired and checked by reading, not compiled, not run on hardware
//
#include <hip/hip_runtime.h>
#include <hip/hip_bf16.h>

// Problem constants (derived from the reference):
//   out[i, s] = sum_k Wa[i, k] * Bt[s, k]
//   M = 512 output channels, K = 3*3*256 = 2304, N = 49 spatial (padded to 64)
#define K_TOT 2304
#define M_TOT 512
#define N_PAD 64

typedef __attribute__((ext_vector_type(16))) __bf16 bf16x16;
typedef __attribute__((ext_vector_type(8)))  float  floatx8;

struct alignas(16) U128 { unsigned int a, b, c, d; };
union FragAB { U128 u[2]; bf16x16 v; };

__device__ __forceinline__ unsigned short f32_to_bf16_rne(float f) {
  unsigned u = __float_as_uint(f);
  u += 0x7FFFu + ((u >> 16) & 1u);   // round-to-nearest-even
  return (unsigned short)(u >> 16);
}

// Kernel 1: W (512,3,3,256) f32, already row-major 512x2304 A-matrix -> bf16.
__global__ void convert_W(const float* __restrict__ W,
                          unsigned short* __restrict__ Wa, int n) {
  int i = blockIdx.x * blockDim.x + threadIdx.x;
  if (i < n) Wa[i] = f32_to_bf16_rne(W[i]);
}

// Kernel 2: build B^T (N_PAD x K_TOT, n-major) in bf16 directly from x.
//   n = h*7 + w  (n >= 49 -> zero pad)
//   k = (d2*3 + d)*256 + l
//   B[n,k] = [0<=h+d2-1<7] * ( [0<=w+d-1<7]  * x[l,     hh, (w +d-2)%7]
//                            + [0<=w2+d-1<7] * x[256+l, hh, (w2+d-2)%7] ),  w2=(w-1)%7
__global__ void build_B(const float* __restrict__ x,
                        unsigned short* __restrict__ Bt) {
  int t = blockIdx.x * blockDim.x + threadIdx.x;
  if (t >= N_PAD * K_TOT) return;
  int n = t / K_TOT;
  int k = t - n * K_TOT;
  float val = 0.f;
  if (n < 49) {
    int h  = n / 7;
    int w  = n - h * 7;
    int d2 = k / 768;
    int rem = k - d2 * 768;
    int d  = rem >> 8;
    int l  = rem & 255;
    int hh = h + d2 - 1;
    if (hh >= 0 && hh < 7) {
      int wa = w + d - 1;                       // first-unfold window index, low 256 ch
      if (wa >= 0 && wa < 7)
        val += x[l * 49 + hh * 7 + ((wa + 6) % 7)];
      int w2 = (w + 6) % 7;                     // high 256 channels shifted by one column
      int wb = w2 + d - 1;
      if (wb >= 0 && wb < 7)
        val += x[(256 + l) * 49 + hh * 7 + ((wb + 6) % 7)];
    }
  }
  Bt[t] = f32_to_bf16_rne(val);
}

// Kernel 3: bf16 WMMA GEMM. One wave per 16x16 D tile.
// grid = 16 blocks: blockIdx = mblk*4 + nblk; 8 waves/block cover 128 M rows.
__global__ void __launch_bounds__(256) wmma_gemm(const unsigned short* __restrict__ Wa,
                                                 const unsigned short* __restrict__ Bt,
                                                 float* __restrict__ out) {
  int wave = threadIdx.x >> 5;
  int lane = threadIdx.x & 31;
  int mblk = blockIdx.x >> 2;        // 0..3
  int nblk = blockIdx.x & 3;         // 0..3
  int i0 = mblk * 128 + wave * 16;   // M tile origin
  int n0 = nblk * 16;                // N tile origin

  // A fragment (16x32 bf16): lane holds row M=lane%16,
  // K chunks [base, base+8) and [base+16, base+24), base = lane<16 ? 0 : 8.
  int mrow  = i0 + (lane & 15);
  int abase = (lane < 16) ? 0 : 8;
  const unsigned short* ap = Wa + mrow * K_TOT + abase;

  // B fragment (32x16 bf16): lane holds col N=lane%16,
  // K chunks [base, base+8) and [base+16, base+24), base = lane<16 ? 0 : 8 (B stored N-major).
  int ncol  = n0 + (lane & 15);
  int bbase = (lane < 16) ? 0 : 8;
  const unsigned short* bp = Bt + ncol * K_TOT + bbase;

  floatx8 acc = {0.f, 0.f, 0.f, 0.f, 0.f, 0.f, 0.f, 0.f};

  #pragma unroll 4
  for (int k0 = 0; k0 < K_TOT; k0 += 32) {
    FragAB a, b;
    a.u[0] = *(const U128*)(ap + k0);        // K base   .. base+7
    a.u[1] = *(const U128*)(ap + k0 + 16);   // K base+16.. base+23
    b.u[0] = *(const U128*)(bp + k0);        // K bbase   .. bbase+7
    b.u[1] = *(const U128*)(bp + k0 + 16);   // K bbase+16.. bbase+23
    acc = __builtin_amdgcn_wmma_f32_16x16x32_bf16(
        /*neg_a=*/false, a.v, /*neg_b=*/false, b.v,
        /*c_mod=*/(short)0, acc, /*reuse_a=*/false, /*reuse_b=*/false);
  }

  // D layout: VGPR r -> M = r + (lane<16 ? 0 : 8), N = lane%16.
  int sc = n0 + (lane & 15);
  if (sc < 49) {
    int ib = i0 + ((lane < 16) ? 0 : 8);
    #pragma unroll
    for (int r = 0; r < 8; ++r)
      out[(ib + r) * 49 + sc] = acc[r];
  }
}

extern "C" void kernel_launch(void* const* d_in, const int* in_sizes, int n_in,
                              void* d_out, int out_size, void* d_ws, size_t ws_size,
                              hipStream_t stream) {
  const float* x = (const float*)d_in[0];   // (1,512,7,7) f32
  const float* W = (const float*)d_in[1];   // (512,3,3,256) f32
  float* out = (float*)d_out;               // (1,512,7,7) f32

  unsigned short* Wa = (unsigned short*)d_ws;                 // 512*2304 bf16 (~2.36 MB)
  unsigned short* Bt = Wa + (size_t)M_TOT * K_TOT;            // 64*2304 bf16 (~295 KB)

  int nW = M_TOT * K_TOT;                                     // 1179648
  convert_W<<<(nW + 255) / 256, 256, 0, stream>>>(W, Wa, nW);
  build_B<<<(N_PAD * K_TOT) / 256, 256, 0, stream>>>(x, Bt);  // 576 blocks
  wmma_gemm<<<16, 256, 0, stream>>>(Wa, Bt, out);
}
